// KNet_Tracker_82592221102170
// MI455X (gfx1250) — matrix operand order repaired and checked by reading, not compiled, not run on hardware
//
#include <hip/hip_runtime.h>
#include <math.h>

typedef __attribute__((ext_vector_type(16))) _Float16 v16h;
typedef __attribute__((ext_vector_type(8)))  float    v8f;

#define PI_F 3.14159265358979323846f
#define NORM_R_SCALE_F (500.0f/150000.0f)

// ---- padded f16 weight layout inside d_ws (element offsets, halves) ----
// W_ih: 768 x 96   @ 0        (73728)
// W_hh: 768 x 256  @ 73728    (196608)
// W1  : 256 x 256  @ 270336   (65536)
// W2  : 128 x 256  @ 335872   (32768)
// W3  : 176 x 128  @ 368640   (22528)   total 391168 halves = 782336 B
#define WS_WIH 0
#define WS_WHH 73728
#define WS_W1  270336
#define WS_W2  335872
#define WS_W3  368640

__device__ __forceinline__ v8f vzero8() {
    v8f z = {0.f,0.f,0.f,0.f,0.f,0.f,0.f,0.f};
    return z;
}

// A-fragment (16-bit, 16x32 MxK) from LDS f16, row stride 'strideH' halves.
// ISA: lane L holds row m=L%16; VGPR j holds K = (j>=4)*16 + (L>=16)*8 + 2*(j&3) + {0,1}.
__device__ __forceinline__ v16h load_a_frag(const _Float16* A, int strideH, int m, int kbase, int lane) {
    union { v16h v; unsigned u32[8]; } u;
    const unsigned* Ar = (const unsigned*)(A + (size_t)m * strideH);
    int kd = (kbase >> 1) + ((lane & 16) ? 4 : 0);
#pragma unroll
    for (int j = 0; j < 8; ++j)
        u.u32[j] = Ar[kd + (j & 3) + ((j >> 2) << 3)];
    return u.v;
}

// B-fragment (16-bit, 32x16 KxN) from pre-converted, padded f16 weights.
// Wrow = base of weight row n (length KpadH). VGPR j holds K = (j>=4)*16 + (L>=16)*8 + 2*(j&3) + {0,1}
// -> two aligned uint4 loads (K 8h..8h+7 and 16+8h..23+8h), no guards.
__device__ __forceinline__ v16h load_b_frag(const _Float16* Wrow, int kbase, int lane) {
    union { v16h v; uint4 q[2]; } u;
    const uint4* p = (const uint4*)(Wrow + kbase + ((lane & 16) ? 8 : 0));
    u.q[0] = p[0];
    u.q[1] = p[2];
    return u.v;
}

// D(16x16,f32) accumulating A(16xK) * W^T tile over NCHUNKS K-chunks of 32.
template<int NCHUNKS>
__device__ __forceinline__ v8f gemm_tile(const _Float16* Alds, int strideH, int row0,
                                         const _Float16* Wh, int KpadH, int ntile, int lane) {
    v8f acc = vzero8();
    int m = row0 + (lane & 15);
    const _Float16* wr = Wh + (size_t)(ntile * 16 + (lane & 15)) * KpadH;
#pragma unroll
    for (int c = 0; c < NCHUNKS; ++c) {
        v16h a = load_a_frag(Alds, strideH, m, c * 32, lane);
        v16h b = load_b_frag(wr, c * 32, lane);
        acc = __builtin_amdgcn_wmma_f32_16x16x32_f16(false, a, false, b, (short)0, acc, false, false);
    }
    return acc;
}

// Fused 3-gate GEMM: one A-fragment feeds weight rows n, n+256, n+512 (GRU r/z/n gates).
template<int NCHUNKS>
__device__ __forceinline__ void gemm3(const _Float16* Alds, int strideH, int row0,
                                      const _Float16* Wh, int KpadH, int ht, int lane,
                                      v8f& d0, v8f& d1, v8f& d2) {
    int m = row0 + (lane & 15);
    const _Float16* w0 = Wh + (size_t)(ht * 16 + (lane & 15)) * KpadH;
    const _Float16* w1 = w0 + (size_t)256 * KpadH;
    const _Float16* w2 = w1 + (size_t)256 * KpadH;
#pragma unroll
    for (int c = 0; c < NCHUNKS; ++c) {
        v16h a = load_a_frag(Alds, strideH, m, c * 32, lane);
        d0 = __builtin_amdgcn_wmma_f32_16x16x32_f16(false, a, false, load_b_frag(w0, c*32, lane), (short)0, d0, false, false);
        d1 = __builtin_amdgcn_wmma_f32_16x16x32_f16(false, a, false, load_b_frag(w1, c*32, lane), (short)0, d1, false, false);
        d2 = __builtin_amdgcn_wmma_f32_16x16x32_f16(false, a, false, load_b_frag(w2, c*32, lane), (short)0, d2, false, false);
    }
}

// ---- prepass: convert f32 weights -> padded f16 (zero-fill K/N tails) ----
__global__ void convert_pad_f16(const float* __restrict__ src, _Float16* __restrict__ dst,
                                int Ksrc, int Nsrc, int Kpad, int total) {
    int i = blockIdx.x * blockDim.x + threadIdx.x;
    if (i >= total) return;
    int n = i / Kpad;
    int k = i - n * Kpad;
    float v = (n < Nsrc && k < Ksrc) ? src[(size_t)n * Ksrc + k] : 0.0f;
    dst[i] = (_Float16)v;
}

__global__ __launch_bounds__(64) void knet_tracker_kernel(
    const float* __restrict__ meas, const float* __restrict__ maskp,
    const float* __restrict__ dtp,  const float* __restrict__ basep,
    const float* __restrict__ xprev, const float* __restrict__ hxg,
    const float* __restrict__ b_ih, const float* __restrict__ b_hh,
    const float* __restrict__ ln_g, const float* __restrict__ ln_b,
    const float* __restrict__ b1,  const float* __restrict__ b2,
    const float* __restrict__ b3,
    const _Float16* __restrict__ wsh,   // padded f16 weights
    float* __restrict__ out, int nB)
{
    // LDS layout (byte offsets), regions reused by phase liveness:
    //   [    0, 3456) innov f32 [32][27]
    //   [ 3456, 4224) x_minus f32 [32][6]
    //   [ 4224,10368) rnn_in f16 [32][96]   -> Km f32 [32][176] (W3..end)
    //   [10368,26752) hx f16 [32][256]      -> h_ln f16 [32][256] (LN..W1)
    //   [26752,59520) h f32 [32][256]       -> f1 f16 [32][256] (W1..W2), f2 f16 [32][128] @43136
    __shared__ alignas(16) unsigned char smem[59520];
    float*    innovS = (float*)(smem + 0);
    float*    xmS    = (float*)(smem + 3456);
    _Float16* rnnS   = (_Float16*)(smem + 4224);
    _Float16* hxS    = (_Float16*)(smem + 10368);
    float*    hS     = (float*)(smem + 26752);
    _Float16* hlnS   = (_Float16*)(smem + 10368);
    _Float16* f1S    = (_Float16*)(smem + 26752);
    _Float16* f2S    = (_Float16*)(smem + 43136);
    float*    kmS    = (float*)(smem + 4224);

    const _Float16* Wih_h = wsh + WS_WIH;
    const _Float16* Whh_h = wsh + WS_WHH;
    const _Float16* W1_h  = wsh + WS_W1;
    const _Float16* W2_h  = wsh + WS_W2;
    const _Float16* W3_h  = wsh + WS_W3;

    const int tid  = threadIdx.x;
    const int lane = tid & 31;     // wave32
    const int wave = tid >> 5;     // 2 waves, each owns a 16-row tile
    const int g0   = blockIdx.x * 32;
    float* xout = out;
    float* hout = out + (size_t)nB * 6;

    // ---- Phase 0: physics prelude + rnn_in packing (f16) ----
    if (tid < 32) {
        int r = tid, g = g0 + r;
        float dt  = dtp[g];
        float dts = dt * NORM_R_SCALE_F;
        float xp[6];
#pragma unroll
        for (int i = 0; i < 6; ++i) xp[i] = xprev[(size_t)g*6 + i];
        float xm[6];
        xm[0] = xp[0] + dts*xp[1]; xm[1] = xp[1];
        xm[2] = xp[2] + dts*xp[3]; xm[3] = xp[3];
        xm[4] = xp[4] + dts*xp[5]; xm[5] = xp[5];
#pragma unroll
        for (int i = 0; i < 6; ++i) xmS[r*6 + i] = xm[i];

        float bl[12];
#pragma unroll
        for (int i = 0; i < 12; ++i) bl[i] = basep[(size_t)g*12 + i];

        float ypred[27];
#pragma unroll
        for (int p = 0; p < 3; ++p) {
            float x0 = xm[0], x2 = xm[2], x4 = xm[4];
            if (p >= 1) { x0 -= bl[(p-1)*6+0]; x2 -= bl[(p-1)*6+2]; x4 -= bl[(p-1)*6+4]; }
            float rxy = sqrtf(x0*x0 + x2*x2 + 1e-9f);
            float az  = atan2f(x2, x0) * (1.0f/PI_F);
            float el  = atan2f(x4, rxy) * (1.0f/PI_F);
            float rr  = sqrtf(x0*x0 + x2*x2 + x4*x4 + 1e-9f);
            ypred[p*9+0]=az; ypred[p*9+1]=el; ypred[p*9+2]=rr;
            ypred[p*9+3]=az; ypred[p*9+4]=el; ypred[p*9+5]=0.f;
            ypred[p*9+6]=az; ypred[p*9+7]=0.f; ypred[p*9+8]=0.f;
        }
#pragma unroll
        for (int k = 0; k < 27; ++k) {
            float inn = meas[(size_t)g*27 + k] - ypred[k];
            int kk = k % 9;
            bool ang = (kk==0)||(kk==1)||(kk==3)||(kk==4)||(kk==6);
            if (ang) {
                float a = inn * PI_F;
                a = a - (2.0f*PI_F) * rintf(a * (0.5f/PI_F));
                inn = a * (1.0f/PI_F);
            }
            float mk = maskp[(size_t)g*27 + k];
            inn *= mk;
            innovS[r*27 + k]    = inn;
            rnnS[r*96 + k]      = (_Float16)inn;
            rnnS[r*96 + 27 + k] = (_Float16)mk;
        }
        rnnS[r*96 + 54] = (_Float16)dt;
#pragma unroll
        for (int i = 0; i < 6;  ++i) rnnS[r*96 + 55 + i] = (_Float16)xm[i];
#pragma unroll
        for (int i = 0; i < 12; ++i) rnnS[r*96 + 61 + i] = (_Float16)bl[i];
#pragma unroll
        for (int i = 73; i < 96; ++i) rnnS[r*96 + i] = (_Float16)0.0f;
    }
    // stage hx (f32 -> f16), coalesced
    for (int i = tid; i < 32*256; i += 64) {
        int r = i >> 8, c = i & 255;
        hxS[i] = (_Float16)hxg[(size_t)(g0 + r)*256 + c];
    }
    __syncthreads();

    // ---- Phase 1: GRU via WMMA (6 accumulators per hidden 16-tile) ----
    {
        int row0 = wave * 16;
        int nb   = lane & 15;
        int hi8  = (lane & 16) ? 8 : 0;
#pragma unroll 1
        for (int ht = 0; ht < 16; ++ht) {
            v8f gir = vzero8(), giz = vzero8(), gin = vzero8();
            v8f ghr = vzero8(), ghz = vzero8(), ghn = vzero8();
            gemm3<3>(rnnS,  96, row0, Wih_h,  96, ht, lane, gir, giz, gin);
            gemm3<8>(hxS,  256, row0, Whh_h, 256, ht, lane, ghr, ghz, ghn);
            int col = ht*16 + nb;
            float bir = b_ih[col], biz = b_ih[256+col], bin = b_ih[512+col];
            float bhr = b_hh[col], bhz = b_hh[256+col], bhn = b_hh[512+col];
#pragma unroll
            for (int rr = 0; rr < 8; ++rr) {
                int row = row0 + rr + hi8;          // C/D layout: lanes 16-31 -> M = r + 8
                int g   = g0 + row;
                float r_g = 1.0f/(1.0f + expf(-(gir[rr]+bir + ghr[rr]+bhr)));
                float z_g = 1.0f/(1.0f + expf(-(giz[rr]+biz + ghz[rr]+bhz)));
                float n_g = tanhf(gin[rr]+bin + r_g*(ghn[rr]+bhn));
                float hxv = (float)hxS[row*256 + col];
                float h   = (1.0f - z_g)*n_g + z_g*hxv;
                hS[row*256 + col] = h;
                hout[(size_t)g*256 + col] = h;       // h_new output (f32)
            }
        }
    }
    __syncthreads();

    // ---- Phase 2: LayerNorm (f32 stats) -> h_ln f16 ----
    if (tid < 32) {
        int r = tid;
        float s = 0.f, s2 = 0.f;
        for (int c = 0; c < 256; ++c) { float v = hS[r*256 + c]; s += v; s2 += v*v; }
        float mu  = s * (1.0f/256.0f);
        float var = s2 * (1.0f/256.0f) - mu*mu;
        float inv = rsqrtf(var + 1e-5f);
        for (int c = 0; c < 256; ++c) {
            float v = (hS[r*256 + c] - mu) * inv * ln_g[c] + ln_b[c];
            hlnS[r*256 + c] = (_Float16)v;
        }
    }
    __syncthreads();

    // ---- Phase 3: f1 = relu(h_ln @ W1^T + b1) ----
    {
        int row0 = wave*16, nb = lane & 15, hi8 = (lane & 16) ? 8 : 0;
#pragma unroll 1
        for (int nt = 0; nt < 16; ++nt) {
            v8f acc = gemm_tile<8>(hlnS, 256, row0, W1_h, 256, nt, lane);
            float bb = b1[nt*16 + nb];
#pragma unroll
            for (int rr = 0; rr < 8; ++rr) {
                float v = acc[rr] + bb;
                v = v > 0.f ? v : 0.f;
                f1S[(row0+rr+hi8)*256 + nt*16 + nb] = (_Float16)v;
            }
        }
    }
    __syncthreads();

    // ---- Phase 4: f2 = relu(f1 @ W2^T + b2) ----
    {
        int row0 = wave*16, nb = lane & 15, hi8 = (lane & 16) ? 8 : 0;
#pragma unroll 1
        for (int nt = 0; nt < 8; ++nt) {
            v8f acc = gemm_tile<8>(f1S, 256, row0, W2_h, 256, nt, lane);
            float bb = b2[nt*16 + nb];
#pragma unroll
            for (int rr = 0; rr < 8; ++rr) {
                float v = acc[rr] + bb;
                v = v > 0.f ? v : 0.f;
                f2S[(row0+rr+hi8)*128 + nt*16 + nb] = (_Float16)v;
            }
        }
    }
    __syncthreads();

    // ---- Phase 5: K = clip(f2 @ W3^T + b3, +-3) -> Km f32 (N padded to 176) ----
    {
        int row0 = wave*16, nb = lane & 15, hi8 = (lane & 16) ? 8 : 0;
#pragma unroll 1
        for (int nt = 0; nt < 11; ++nt) {
            v8f acc = gemm_tile<4>(f2S, 128, row0, W3_h, 128, nt, lane);
            int n = nt*16 + nb;
            float bb = (n < 162) ? b3[n] : 0.0f;
#pragma unroll
            for (int rr = 0; rr < 8; ++rr) {
                float v = acc[rr] + bb;
                v = fminf(fmaxf(v, -3.0f), 3.0f);
                kmS[(row0+rr+hi8)*176 + n] = v;
            }
        }
    }
    __syncthreads();

    // ---- Phase 6: x_new = clip(x_minus + K @ innov, +-5) ----
    if (tid < 32) {
        int r = tid, g = g0 + r;
#pragma unroll
        for (int s = 0; s < 6; ++s) {
            float c = 0.f;
#pragma unroll
            for (int m = 0; m < 27; ++m)
                c += kmS[r*176 + s*27 + m] * innovS[r*27 + m];
            float x = xmS[r*6 + s] + c;
            x = fminf(fmaxf(x, -5.0f), 5.0f);
            xout[(size_t)g*6 + s] = x;
        }
    }
}

extern "C" void kernel_launch(void* const* d_in, const int* in_sizes, int n_in,
                              void* d_out, int out_size, void* d_ws, size_t ws_size,
                              hipStream_t stream) {
    const float* meas  = (const float*)d_in[0];
    const float* maskp = (const float*)d_in[1];
    const float* dtp   = (const float*)d_in[2];
    const float* basep = (const float*)d_in[3];
    const float* xprev = (const float*)d_in[4];
    const float* hxg   = (const float*)d_in[5];
    const float* W_ih  = (const float*)d_in[6];
    const float* W_hh  = (const float*)d_in[7];
    const float* b_ih  = (const float*)d_in[8];
    const float* b_hh  = (const float*)d_in[9];
    const float* ln_g  = (const float*)d_in[10];
    const float* ln_b  = (const float*)d_in[11];
    const float* W1    = (const float*)d_in[12];
    const float* b1    = (const float*)d_in[13];
    const float* W2    = (const float*)d_in[14];
    const float* b2    = (const float*)d_in[15];
    const float* W3    = (const float*)d_in[16];
    const float* b3    = (const float*)d_in[17];

    _Float16* wsh = (_Float16*)d_ws;

    // prepass: convert + pad weights to f16 in workspace
    convert_pad_f16<<<(73728  + 255)/256, 256, 0, stream>>>(W_ih, wsh + WS_WIH,  73, 768,  96, 73728);
    convert_pad_f16<<<(196608 + 255)/256, 256, 0, stream>>>(W_hh, wsh + WS_WHH, 256, 768, 256, 196608);
    convert_pad_f16<<<(65536  + 255)/256, 256, 0, stream>>>(W1,   wsh + WS_W1,  256, 256, 256, 65536);
    convert_pad_f16<<<(32768  + 255)/256, 256, 0, stream>>>(W2,   wsh + WS_W2,  256, 128, 256, 32768);
    convert_pad_f16<<<(22528  + 255)/256, 256, 0, stream>>>(W3,   wsh + WS_W3,  128, 162, 128, 22528);

    int nB = in_sizes[2];            // dt is [B]
    int blocks = nB / 32;            // 32 rows per block (2 waves x 16-row tiles)
    knet_tracker_kernel<<<blocks, 64, 0, stream>>>(
        meas, maskp, dtp, basep, xprev, hxg,
        b_ih, b_hh, ln_g, ln_b, b1, b2, b3,
        wsh, (float*)d_out, nB);
}
